// MultiHeadAttention_76871324664580
// MI455X (gfx1250) — matrix operand order repaired and checked by reading, not compiled, not run on hardware
//
#include <hip/hip_runtime.h>
#include <hip/hip_bf16.h>

#define HEADS   16
#define D_MODEL 1024
#define D_H     64
#define BATCH   4
#define SEQ     2048
#define MROWS   (BATCH * SEQ)   // 8192

typedef __attribute__((ext_vector_type(16))) __bf16 v16bf;
typedef __attribute__((ext_vector_type(8)))  __bf16 v8bf;
typedef __attribute__((ext_vector_type(8)))  float  v8f;
typedef int v4si __attribute__((vector_size(16)));

// ---------------------------------------------------------------------------
// CDNA5 async global->LDS staging (ASYNCcnt-tracked); guarded fallback.
// Builtin signature (from hipcc diagnostic): takes pointers to int __vector(4)
// with no explicit address space -> pass generic v4si* for both operands.
// ---------------------------------------------------------------------------
#if __has_builtin(__builtin_amdgcn_global_load_async_to_lds_b128)
#define HAS_ASYNC_LDS 1
#endif

static __device__ inline void async_copy_b128(const void* gsrc, void* ldst) {
#ifdef HAS_ASYNC_LDS
  __builtin_amdgcn_global_load_async_to_lds_b128((v4si*)gsrc, (v4si*)ldst, 0, 0);
#else
  *(v8bf*)ldst = *(const v8bf*)gsrc;   // sync fallback: load + ds_store
#endif
}

static __device__ inline void wait_async0() {
#ifdef HAS_ASYNC_LDS
#if __has_builtin(__builtin_amdgcn_s_wait_asynccnt)
  __builtin_amdgcn_s_wait_asynccnt(0);
#else
  asm volatile("s_wait_asynccnt 0x0" ::: "memory");
#endif
#endif
}

// ---------------------------------------------------------------------------
// WMMA wrappers: D(16x16 f32) = A(16x32 bf16) * B(32x16 bf16) + C
// ---------------------------------------------------------------------------
static __device__ inline v8f wmma_bf16(v16bf a, v16bf b, v8f c) {
  return __builtin_amdgcn_wmma_f32_16x16x32_bf16(false, a, false, b,
                                                 (short)0, c, false, false);
}
// reuse_a hint: legal because it is issued back-to-back after an identical
// WMMA with the same A operand (ISA 7.12 OPSEL.RA).
static __device__ inline v8f wmma_bf16_ra(v16bf a, v16bf b, v8f c) {
  return __builtin_amdgcn_wmma_f32_16x16x32_bf16(false, a, false, b,
                                                 (short)0, c, true, false);
}

// ---------------------------------------------------------------------------
// Fragment loaders (wave32): A and B share one contiguous k order, not the ISA table order.
// A (16x32, 16-bit): lane = row (lane&15), kbase = (lane>>4)*16;
//   element e -> K = kbase+e (contiguous, same order as B)
// B (32x16, 16-bit): lane col n = lane&15, kgroup = (lane>>4)*16;
//   element e -> K = kgroup+e (contiguous along K for a fixed column)
// C/D (16x16, f32): lane holds col n = lane&15, rows M = r + 8*(lane>>4)
// ---------------------------------------------------------------------------
static __device__ inline v16bf load_a_f32(const float* X, int row0, int ld,
                                          int k0, int lane) {
  int row = row0 + (lane & 15);
  int kb  = k0 + ((lane >> 4) << 4);
  const float* p = X + (size_t)row * ld + kb;
  v16bf a;
#pragma unroll
  for (int e = 0; e < 8; ++e) a[e] = (__bf16)p[e];
#pragma unroll
  for (int e = 0; e < 8; ++e) a[8 + e] = (__bf16)p[8 + e];
  return a;
}

static __device__ inline v16bf load_a_bf16(const __bf16* X, int row0, int ld,
                                           int k0, int lane) {
  int row = row0 + (lane & 15);
  int kb  = k0 + ((lane >> 4) << 4);
  const __bf16* p = X + (size_t)row * ld + kb;
  v8bf lo = *(const v8bf*)p;
  v8bf hv = *(const v8bf*)(p + 8);
  return __builtin_shufflevector(lo, hv, 0, 1, 2, 3, 4, 5, 6, 7,
                                         8, 9, 10, 11, 12, 13, 14, 15);
}

// B fragment from a row-major [n][k] source: one 32B contiguous read per lane.
static __device__ inline v16bf load_b_bf16(const __bf16* Wnk, int n0, int ld,
                                           int k0, int lane) {
  int n  = n0 + (lane & 15);
  int kg = k0 + ((lane >> 4) << 4);
  return *(const v16bf*)(Wnk + (size_t)n * ld + kg);
}

// B fragment from a staged LDS tile (tile is [64 n][32 k] bf16)
static __device__ inline v16bf load_b_lds(const unsigned short* tile, int nsub,
                                          int lane) {
  int n  = nsub + (lane & 15);
  int kg = (lane >> 4) << 4;
  return *(const v16bf*)((const __bf16*)tile + n * 32 + kg);
}

static __device__ inline unsigned short f2bfbits(float f) {
  __bf16 h = (__bf16)f;
  return __builtin_bit_cast(unsigned short, h);
}

// half-wave (16 lane) reductions: xor masks 1,2,4,8 stay inside each 16-group
static __device__ inline float half_max(float x) {
  x = fmaxf(x, __shfl_xor(x, 1));
  x = fmaxf(x, __shfl_xor(x, 2));
  x = fmaxf(x, __shfl_xor(x, 4));
  x = fmaxf(x, __shfl_xor(x, 8));
  return x;
}
static __device__ inline float half_sum(float x) {
  x += __shfl_xor(x, 1);
  x += __shfl_xor(x, 2);
  x += __shfl_xor(x, 4);
  x += __shfl_xor(x, 8);
  return x;
}

// ---------------------------------------------------------------------------
// Stage 0: weights fp32 -> bf16 (re-read 8192x afterwards; convert once)
// ---------------------------------------------------------------------------
__global__ __launch_bounds__(256)
void convert_w_kernel(const float* Wq, const float* Wk, const float* Wv,
                      const float* Wo, __bf16* wbf) {
  int w = blockIdx.y;
  const float* src = (w == 0) ? Wq : (w == 1) ? Wk : (w == 2) ? Wv : Wo;
  __bf16* dst = wbf + (size_t)w * D_MODEL * D_MODEL;
  size_t idx = ((size_t)blockIdx.x * 256 + threadIdx.x) * 16;
#pragma unroll
  for (int e = 0; e < 16; ++e) dst[idx + e] = (__bf16)src[idx + e];
}

// ---------------------------------------------------------------------------
// Stage 1: QKV projection  out[m,n] = sum_k X[m,k]*W[n,k] + b[n]
// grid (64, 16, 3), block 256 (8 waves). Wave tile: 16(M) x 64(N).
// The 64x32 bf16 weight tile is shared by all 8 waves -> async-staged into
// LDS (double-buffered), cutting B-side traffic 8x. One barrier per k-step.
// z=0/1 -> store [b,h,s,d]; z=2 -> store V transposed [b,h,d,s]
// ---------------------------------------------------------------------------
__global__ __launch_bounds__(256)
void qkv_proj_kernel(const float* q, const float* k, const float* v,
                     const float* bq, const float* bk, const float* bv,
                     const __bf16* wbf, __bf16* qh, __bf16* kh, __bf16* vt) {
  __shared__ unsigned short btile[2][64 * 32];   // 2 x 4KB

  const int z = blockIdx.z;
  const float* X    = (z == 0) ? q  : (z == 1) ? k  : v;
  const float* bias = (z == 0) ? bq : (z == 1) ? bk : bv;
  const __bf16* W = wbf + (size_t)z * D_MODEL * D_MODEL;

  const int tid  = threadIdx.x;
  const int lane = tid & 31;
  const int wave = tid >> 5;
  const int m0   = (blockIdx.x * 8 + wave) * 16;   // 0..8176
  const int n0   = blockIdx.y * 64;
  const int col  = lane & 15;
  const int lh   = lane >> 4;

  // staging slot for this thread: row sn (of 64), 8-elem k-chunk sk
  const int sn = tid >> 2, sk = (tid & 3) * 8;
  const __bf16* wsrc = W + (size_t)(n0 + sn) * D_MODEL + sk;

  v8f acc[4] = {};
  async_copy_b128(wsrc, &btile[0][sn * 32 + sk]);   // prologue stage k0=0
  int buf = 0;
  for (int k0 = 0; k0 < D_MODEL; k0 += 32) {
    wait_async0();
    __syncthreads();                                // tile[buf] ready, block-wide
    if (k0 + 32 < D_MODEL)
      async_copy_b128(wsrc + (k0 + 32), &btile[buf ^ 1][sn * 32 + sk]);

    v16bf afrag = load_a_f32(X, m0, D_MODEL, k0, lane);
    acc[0] = wmma_bf16   (afrag, load_b_lds(btile[buf],  0, lane), acc[0]);
    acc[1] = wmma_bf16_ra(afrag, load_b_lds(btile[buf], 16, lane), acc[1]);
    acc[2] = wmma_bf16_ra(afrag, load_b_lds(btile[buf], 32, lane), acc[2]);
    acc[3] = wmma_bf16_ra(afrag, load_b_lds(btile[buf], 48, lane), acc[3]);
    buf ^= 1;
  }

#pragma unroll
  for (int c = 0; c < 4; ++c) {
    const int n  = n0 + c * 16 + col;
    const float bn = bias[n];
    const int h = n >> 6, d = n & 63;
    if (z < 2) {
      __bf16* out = (z == 0) ? qh : kh;
#pragma unroll
      for (int r = 0; r < 8; ++r) {
        int m = m0 + r + 8 * lh;
        int b = m >> 11, s = m & (SEQ - 1);
        out[(((size_t)(b * HEADS + h) * SEQ + s) << 6) + d] =
            (__bf16)(acc[c][r] + bn);
      }
    } else {
      int mb = m0 + 8 * lh;
      int b = mb >> 11, sbase = mb & (SEQ - 1);
      size_t base = ((size_t)(b * HEADS + h) * D_H + d) * SEQ + sbase;
      v8bf pack;
#pragma unroll
      for (int r = 0; r < 8; ++r) pack[r] = (__bf16)(acc[c][r] + bn);
      *(v8bf*)(vt + base) = pack;   // 16B aligned (sbase % 8 == 0)
    }
  }
}

// ---------------------------------------------------------------------------
// Stage 2: flash attention. grid (16, 64), block 256 (8 waves).
// Each wave: one 16-row Q tile of one (b,h); loops 32 keys/iter (8 WMMAs).
// K/V (32 MB bf16 total) are L2-resident; fragments are single 32B loads,
// with prefetch of the next key tile to cover L2 latency.
// ---------------------------------------------------------------------------
__global__ __launch_bounds__(256)
void attn_kernel(const __bf16* qh, const __bf16* kh, const __bf16* vt,
                 __bf16* ctx) {
  __shared__ unsigned short plds[8 * 16 * 32];   // 1KB per wave

  const int lane = threadIdx.x & 31;
  const int wave = threadIdx.x >> 5;
  const int bh = blockIdx.y;
  const int b = bh >> 4, h = bh & 15;
  const int s0 = (blockIdx.x * 8 + wave) * 16;
  const int col = lane & 15;
  const int lh  = lane >> 4;

  const __bf16* Q = qh + (size_t)bh * SEQ * D_H;
  const __bf16* K = kh + (size_t)bh * SEQ * D_H;
  const __bf16* V = vt + (size_t)bh * D_H * SEQ;   // [d][s]
  unsigned short* myl = plds + wave * 512;

  const v16bf qa = load_a_bf16(Q, s0, D_H, 0, lane);
  const v16bf qb = load_a_bf16(Q, s0, D_H, 32, lane);

  const float SCALE = 0.125f;   // 1/sqrt(64)
  float mrow[8], lrow[8];
  v8f acc[4] = {};
#pragma unroll
  for (int r = 0; r < 8; ++r) { mrow[r] = -1e30f; lrow[r] = 0.0f; }

  for (int t = 0; t < SEQ; t += 32) {
    // prefetch next key tile rows (global_prefetch_b8)
    if (t + 32 < SEQ) {
      __builtin_prefetch(K + (size_t)(t + 32 + col) * D_H, 0, 3);
      __builtin_prefetch(V + (size_t)(col * 4) * SEQ + t + 32, 0, 3);
    }

    // ---- logits: two 16x16 tiles for keys [t,t+16) and [t+16,t+32) ----
    v16bf k0a = load_b_bf16(K, t,      D_H, 0,  lane);
    v16bf k0b = load_b_bf16(K, t,      D_H, 32, lane);
    v16bf k1a = load_b_bf16(K, t + 16, D_H, 0,  lane);
    v16bf k1b = load_b_bf16(K, t + 16, D_H, 32, lane);
    v8f sc0 = {};
    sc0 = wmma_bf16(qa, k0a, sc0);
    sc0 = wmma_bf16(qb, k0b, sc0);
    v8f sc1 = {};
    sc1 = wmma_bf16(qa, k1a, sc1);
    sc1 = wmma_bf16(qb, k1b, sc1);

    // ---- online softmax update ----
    float alpha[8];
#pragma unroll
    for (int r = 0; r < 8; ++r) {
      float a = sc0[r] * SCALE, c = sc1[r] * SCALE;
      sc0[r] = a; sc1[r] = c;
      float tm = half_max(fmaxf(a, c));
      float mn = fmaxf(mrow[r], tm);
      alpha[r] = __expf(mrow[r] - mn);
      mrow[r] = mn;
    }
    // P -> LDS (C layout rows -> 16x32 bf16 tile), plus running row sums
#pragma unroll
    for (int r = 0; r < 8; ++r) {
      float p0 = __expf(sc0[r] - mrow[r]);
      float p1 = __expf(sc1[r] - mrow[r]);
      lrow[r] = lrow[r] * alpha[r] + half_sum(p0 + p1);
      myl[(r + 8 * lh) * 32 + col]      = f2bfbits(p0);
      myl[(r + 8 * lh) * 32 + 16 + col] = f2bfbits(p1);
    }
    asm volatile("s_wait_dscnt 0" ::: "memory");   // wave-local LDS transpose

    // re-read P in A-matrix layout
    v16bf pfrag;
    {
      const __bf16* pl = reinterpret_cast<const __bf16*>(myl);
      int row = lane & 15;
      int kb  = (lane >> 4) << 4;
      v8bf lo = *(const v8bf*)(pl + row * 32 + kb);
      v8bf hv = *(const v8bf*)(pl + row * 32 + kb + 8);
      pfrag = __builtin_shufflevector(lo, hv, 0, 1, 2, 3, 4, 5, 6, 7,
                                              8, 9, 10, 11, 12, 13, 14, 15);
    }

    // ---- ctx accumulate: O[16x64] += P(16x32) @ V(32x64) ----
#pragma unroll
    for (int c = 0; c < 4; ++c) {
#pragma unroll
      for (int r = 0; r < 8; ++r) acc[c][r] *= alpha[r];
      v16bf vf = load_b_bf16(V, c * 16, SEQ, t, lane);  // B[k=t..][n=d]
      acc[c] = wmma_bf16(pfrag, vf, acc[c]);
    }
  }

  // ---- finalize: O /= l, store ctx as [b, s, h*64+d] bf16 ----
  float inv[8];
#pragma unroll
  for (int r = 0; r < 8; ++r) inv[r] = 1.0f / lrow[r];
#pragma unroll
  for (int c = 0; c < 4; ++c) {
#pragma unroll
    for (int r = 0; r < 8; ++r) {
      int s = s0 + r + 8 * lh;
      ctx[(size_t)(b * SEQ + s) * D_MODEL + h * 64 + c * 16 + col] =
          (__bf16)(acc[c][r] * inv[r]);
    }
  }
}

// ---------------------------------------------------------------------------
// Stage 3: output projection  out[m,n] = sum_k ctx[m,k]*Wo[n,k] + bo[n] (f32)
// Same async-staged-B structure as stage 1.
// ---------------------------------------------------------------------------
__global__ __launch_bounds__(256)
void out_proj_kernel(const __bf16* ctx, const __bf16* Wo, const float* bo,
                     float* out) {
  __shared__ unsigned short btile[2][64 * 32];

  const int tid  = threadIdx.x;
  const int lane = tid & 31;
  const int wave = tid >> 5;
  const int m0 = (blockIdx.x * 8 + wave) * 16;
  const int n0 = blockIdx.y * 64;
  const int col = lane & 15;
  const int lh  = lane >> 4;

  const int sn = tid >> 2, sk = (tid & 3) * 8;
  const __bf16* wsrc = Wo + (size_t)(n0 + sn) * D_MODEL + sk;

  v8f acc[4] = {};
  async_copy_b128(wsrc, &btile[0][sn * 32 + sk]);
  int buf = 0;
  for (int k0 = 0; k0 < D_MODEL; k0 += 32) {
    wait_async0();
    __syncthreads();
    if (k0 + 32 < D_MODEL)
      async_copy_b128(wsrc + (k0 + 32), &btile[buf ^ 1][sn * 32 + sk]);

    v16bf a = load_a_bf16(ctx, m0, D_MODEL, k0, lane);
    acc[0] = wmma_bf16   (a, load_b_lds(btile[buf],  0, lane), acc[0]);
    acc[1] = wmma_bf16_ra(a, load_b_lds(btile[buf], 16, lane), acc[1]);
    acc[2] = wmma_bf16_ra(a, load_b_lds(btile[buf], 32, lane), acc[2]);
    acc[3] = wmma_bf16_ra(a, load_b_lds(btile[buf], 48, lane), acc[3]);
    buf ^= 1;
  }
#pragma unroll
  for (int c = 0; c < 4; ++c) {
    int n = n0 + c * 16 + col;
    float bias = bo[n];
#pragma unroll
    for (int r = 0; r < 8; ++r) {
      out[(size_t)(m0 + r + 8 * lh) * D_MODEL + n] = acc[c][r] + bias;
    }
  }
}

// ---------------------------------------------------------------------------
// Workspace layout (bytes); total 72 MB
// ---------------------------------------------------------------------------
#define WBF_OFF ((size_t)0)                         // 4 * 1M bf16 = 8 MB
#define QH_OFF  ((size_t)8388608)                   // 16 MB  [b,h,s,d]
#define KH_OFF  ((size_t)25165824)                  // 16 MB  [b,h,s,d]
#define VT_OFF  ((size_t)41943040)                  // 16 MB  [b,h,d,s]
#define CTX_OFF ((size_t)58720256)                  // 16 MB  [m, d_model]

extern "C" void kernel_launch(void* const* d_in, const int* in_sizes, int n_in,
                              void* d_out, int out_size, void* d_ws,
                              size_t ws_size, hipStream_t stream) {
  const float* q  = (const float*)d_in[0];
  const float* k  = (const float*)d_in[1];
  const float* v  = (const float*)d_in[2];
  const float* Wq = (const float*)d_in[3];
  const float* bq = (const float*)d_in[4];
  const float* Wk = (const float*)d_in[5];
  const float* bk = (const float*)d_in[6];
  const float* Wv = (const float*)d_in[7];
  const float* bv = (const float*)d_in[8];
  const float* Wo = (const float*)d_in[9];
  const float* bo = (const float*)d_in[10];

  char* ws = (char*)d_ws;
  __bf16* wbf = (__bf16*)(ws + WBF_OFF);
  __bf16* qh  = (__bf16*)(ws + QH_OFF);
  __bf16* kh  = (__bf16*)(ws + KH_OFF);
  __bf16* vt  = (__bf16*)(ws + VT_OFF);
  __bf16* ctx = (__bf16*)(ws + CTX_OFF);
  float* out  = (float*)d_out;

  convert_w_kernel<<<dim3(256, 4), 256, 0, stream>>>(Wq, Wk, Wv, Wo, wbf);
  qkv_proj_kernel<<<dim3(64, 16, 3), 256, 0, stream>>>(q, k, v, bq, bk, bv,
                                                       wbf, qh, kh, vt);
  attn_kernel<<<dim3(16, 64), 256, 0, stream>>>(qh, kh, vt, ctx);
  out_proj_kernel<<<dim3(64, 16), 256, 0, stream>>>(
      ctx, wbf + (size_t)3 * D_MODEL * D_MODEL, bo, out);
}
